// GD_13907104105202
// MI455X (gfx1250) — matrix operand order repaired and checked by reading, not compiled, not run on hardware
//
#include <hip/hip_runtime.h>

// GD unrolled solver, MI455X (gfx1250, wave32, WMMA).
// One 1024-thread workgroup (32 wave32s) per batch element.
// K-split layout: wave wv owns rows [16*(wv&15), +16) x cols [256*(wv>>4), +256)
// of W, held as 8 bf16 WMMA B-fragments = 64 VGPRs -> fits the per-wave VGPR
// budget at 8 waves/SIMD, so W is read from HBM exactly once (256 MB total
// ~= 11 us at 23.3 TB/s) and stays register-resident for all 20 steps.
// x lives in LDS (fp32 recurrence + packed-bf16 mirror for the WMMA A operand);
// the two K-half partial products are summed via an LDS staging buffer.

typedef __attribute__((ext_vector_type(16))) __bf16        v16bf;
typedef __attribute__((ext_vector_type(8)))  float         v8f;
typedef __attribute__((ext_vector_type(4)))  unsigned int  v4u;

#define NN      512
#define KSTEPS  20

static __device__ __forceinline__ unsigned short f2bf(float f) {
    // round-to-nearest-even fp32 -> bf16
    unsigned u = __builtin_bit_cast(unsigned, f);
    u += 0x7FFFu + ((u >> 16) & 1u);
    return (unsigned short)(u >> 16);
}

union AB32 {                 // 32-byte fragment viewed as 8 dwords or v16bf
    struct { v4u lo, hi; } u;
    v16bf f;
};

__global__ __launch_bounds__(1024) void gd_unrolled_wmma(
    const float* __restrict__ W, const float* __restrict__ Bv,
    const float* __restrict__ Sp, float* __restrict__ Out)
{
    __shared__ __align__(16) float          x_f[NN];        // fp32 state
    __shared__ __align__(16) unsigned short x_bf[NN];       // bf16 mirror (A operand)
    __shared__ __align__(16) float          b_s[NN];
    __shared__ __align__(16) float          y_part[2][NN];  // K-half partial sums

    const int tid    = threadIdx.x;
    const int wv     = tid >> 5;         // wave id 0..31
    const int lane   = tid & 31;
    const int col    = lane & 15;        // B column == output row within tile
    const int khalf  = lane >> 4;        // 0: K 0..15 of tile, 1: K 16..31
    const int rowgrp = wv & 15;          // 16 row-groups
    const int ksplit = wv >> 4;          // 0: k 0..255, 1: k 256..511
    const int batch  = blockIdx.x;

    if (tid < NN) {
        b_s[tid]  = Bv[(size_t)batch * NN + tid];
        x_f[tid]  = 0.0f;                // x0 = 0
        x_bf[tid] = 0;
    }
    const float s = Sp[0];

    // ---- Load this wave's 16x256 slab of W as 8 bf16 B-fragments. ----
    // 32x16 bf16 B layout used here (contiguous k): lane n (0-15) = column n, elem e -> K = e;
    // lanes 16-31 = column n, K = 16+e.  Column n of B = W row (16*rowgrp+n),
    // so each lane reads 16 contiguous floats per k-tile (global_load_b128 x4).
    const int row = (rowgrp << 4) + col;
    const float* Wrow = W + ((size_t)batch * NN + row) * NN + (ksplit << 8);
    v16bf Bfrag[8];
#pragma unroll
    for (int t = 0; t < 8; ++t) {
        const float4* p = (const float4*)(Wrow + 32 * t + 16 * khalf);
        float4 f0 = p[0], f1 = p[1], f2 = p[2], f3 = p[3];
        AB32 c;
        c.u.lo.x = f2bf(f0.x) | ((unsigned)f2bf(f0.y) << 16);
        c.u.lo.y = f2bf(f0.z) | ((unsigned)f2bf(f0.w) << 16);
        c.u.lo.z = f2bf(f1.x) | ((unsigned)f2bf(f1.y) << 16);
        c.u.lo.w = f2bf(f1.z) | ((unsigned)f2bf(f1.w) << 16);
        c.u.hi.x = f2bf(f2.x) | ((unsigned)f2bf(f2.y) << 16);
        c.u.hi.y = f2bf(f2.z) | ((unsigned)f2bf(f2.w) << 16);
        c.u.hi.z = f2bf(f3.x) | ((unsigned)f2bf(f3.y) << 16);
        c.u.hi.w = f2bf(f3.z) | ((unsigned)f2bf(f3.w) << 16);
        Bfrag[t] = c.f;
    }

    __syncthreads();

    // A-fragment (x^T in row 0 of a 16x32 bf16 tile):
    // lane 0 carries K {0..15}, lane 16 carries K {16..31};
    // other lanes hold zeros (masked with AND, EXEC stays all-1 for WMMA).
    const v4u* xb = (const v4u*)x_bf;                       // 8 bf16 per v4u
    const unsigned amask = (col == 0) ? 0xFFFFFFFFu : 0u;   // lanes 0 and 16
    const int tbase = (ksplit << 5) + 2 * khalf;            // v4u index base

#pragma unroll 1
    for (int k = 0; k < KSTEPS; ++k) {
        v8f acc = {};                                       // fresh partial y
#pragma unroll
        for (int t = 0; t < 8; ++t) {
            const int idx = tbase + 4 * t;                  // (256*ksplit+32t+16*khalf)/8
            v4u lo = xb[idx];                               // ds_load_b128
            v4u hi = xb[idx + 1];                           // K +8 chunk
            lo &= amask;
            hi &= amask;
            AB32 a; a.u.lo = lo; a.u.hi = hi;
            acc = __builtin_amdgcn_wmma_f32_16x16x32_bf16(
                false, a.f, false, Bfrag[t], (short)0, acc, false, false);
        }
        // Partial y_i (i = 16*rowgrp + n) sits in acc[0] of lane n (D row 0).
        if (lane < 16) y_part[ksplit][(rowgrp << 4) + col] = acc[0];
        __syncthreads();                                    // partials visible, x reads done
        if (tid < NN) {
            const float y  = y_part[0][tid] + y_part[1][tid];
            const float xn = x_f[tid] - s * (y - b_s[tid]);
            x_f[tid]  = xn;
            x_bf[tid] = f2bf(xn);
        }
        __syncthreads();                                    // x update visible
    }

    if (tid < NN) Out[(size_t)batch * NN + tid] = x_f[tid];
}

extern "C" void kernel_launch(void* const* d_in, const int* in_sizes, int n_in,
                              void* d_out, int out_size, void* d_ws, size_t ws_size,
                              hipStream_t stream) {
    const float* w = (const float*)d_in[0];
    const float* b = (const float*)d_in[1];
    const float* s = (const float*)d_in[2];
    float* out = (float*)d_out;
    const int batches = in_sizes[1] / NN;                   // 256
    gd_unrolled_wmma<<<dim3(batches), dim3(1024), 0, stream>>>(w, b, s, out);
}
